// MF_GNN_ActorCritic_7980049236592
// MI455X (gfx1250) — matrix operand order repaired and checked by reading, not compiled, not run on hardware
//
#include <hip/hip_runtime.h>
#include <hip/hip_bf16.h>

// Problem constants (match reference)
#define N_NODES 50000
#define N_EDGES 800000
#define N_BATCH 64
#define FIN     128
#define HIDC    128      // per-head channels
#define EDIM    64
#define NACT    32
#define NHEAD   4
#define HOUT    (NHEAD*HIDC)            // 512
#define ET      (N_EDGES + N_NODES)     // edges incl. self loops

typedef __attribute__((ext_vector_type(16))) _Float16 v16h;
typedef __attribute__((ext_vector_type(8)))  _Float16 v8h;
typedef __attribute__((ext_vector_type(8)))  float    v8f;

// ---------------------------------------------------------------------------
// WMMA fragment loaders (CDNA5 16-bit layouts, wave32)
// A 16x32 (MxK): lanes 0-15 hold K {0..7,16..23}; lanes 16-31 K {8..15,24..31}
// B 32x16 (KxN): lanes 0-15 hold K {0..7,16..23}; lanes 16-31 K {8..15,24..31}
// ---------------------------------------------------------------------------
__device__ __forceinline__ v16h load_a_frag(const _Float16* __restrict__ A,
                                            int ld, int r0, int k0, int lane) {
  int hi = lane >> 4;
  const _Float16* p = A + (size_t)(r0 + (lane & 15)) * ld + k0 + hi * 8;
  v8h lo = *(const v8h*)p;
  v8h hh = *(const v8h*)(p + 16);
  v16h out;
#pragma unroll
  for (int i = 0; i < 8; ++i) { out[i] = lo[i]; out[8 + i] = hh[i]; }
  return out;
}

// C = A[M,K] @ Bt[Nn,K]^T (f16 in, f32 out), optional bias + relu.
// block = 256 threads = 8 waves covering a 128-row x (16*NT)-col tile.
// The shared B strip (rows n0..n0+16*NT-1 of Bt, K-contiguous) is staged
// once into LDS by the whole block; per-wave B fragments then come from
// ds_load instead of redundant global loads. LDS rows padded (+8 halves)
// to avoid 16-way bank conflicts (K*2B is a multiple of the bank span).
template <int NT, int K>
__global__ void __launch_bounds__(256)
wmma_gemm_lds(const _Float16* __restrict__ A, const _Float16* __restrict__ Bt,
              float* __restrict__ C, const float* __restrict__ bias,
              int M, int Nn, int relu) {
  constexpr int LDK = K + 8;                      // padded LDS row stride (halves)
  __shared__ _Float16 bs[NT * 16 * LDK];
  int lane = threadIdx.x & 31;
  int wave = threadIdx.x >> 5;
  int m0 = (blockIdx.x * 8 + wave) * 16;
  int n0 = blockIdx.y * (16 * NT);

  // cooperative stage of B strip into LDS (uint4 = 8 halves per thread-step)
  {
    constexpr int CH = NT * 16 * (K / 8);
    const uint4* src = (const uint4*)(Bt + (size_t)n0 * K);
    for (int i = threadIdx.x; i < CH; i += 256) {
      int row = i / (K / 8), w = i - row * (K / 8);
      *(uint4*)(bs + row * LDK + w * 8) = src[(size_t)row * (K / 8) + w];
    }
  }
  __syncthreads();

  if (m0 < M) {                                   // wave-uniform guard
    int n = lane & 15, hi = lane >> 4;
    v8f acc[NT];
#pragma unroll
    for (int t = 0; t < NT; ++t) acc[t] = (v8f){};
#pragma unroll
    for (int k0 = 0; k0 < K; k0 += 32) {
      v16h a = load_a_frag(A, K, m0, k0, lane);
#pragma unroll
      for (int t = 0; t < NT; ++t) {
        const _Float16* p = bs + (16 * t + n) * LDK + k0 + hi * 8;
        v8h lo = *(const v8h*)p;
        v8h hh = *(const v8h*)(p + 16);
        v16h b;
#pragma unroll
        for (int i = 0; i < 8; ++i) { b[i] = lo[i]; b[8 + i] = hh[i]; }
        acc[t] = __builtin_amdgcn_wmma_f32_16x16x32_f16(false, a, false, b,
                                                        (short)0, acc[t], false, false);
      }
    }
#pragma unroll
    for (int t = 0; t < NT; ++t) {
      int col = n0 + 16 * t + n;
      float bv = bias ? bias[col] : 0.0f;
#pragma unroll
      for (int r = 0; r < 8; ++r) {
        int row = m0 + r + hi * 8;                // C layout: VGPR r -> M=r / M=r+8
        float v = acc[t][r] + bv;
        if (relu) v = fmaxf(v, 0.0f);
        C[(size_t)row * Nn + col] = v;
      }
    }
  }
}

// ---------------------------------------------------------------------------
// Utility / elementwise kernels
// ---------------------------------------------------------------------------
__device__ __forceinline__ void atomicMaxFloat(float* addr, float val) {
  int* a = (int*)addr;
  int old = __float_as_int(*addr);
  while (__int_as_float(old) < val) {
    int assumed = old;
    old = atomicCAS(a, assumed, __float_as_int(val));
    if (old == assumed) break;
  }
}

__global__ void fill_f32(float* p, float v, int n) {
  int i = blockIdx.x * blockDim.x + threadIdx.x;
  if (i < n) p[i] = v;
}
__global__ void cvt_f32_f16(const float* __restrict__ s, _Float16* __restrict__ d, int n) {
  int i = blockIdx.x * blockDim.x + threadIdx.x;
  if (i < n) d[i] = (_Float16)s[i];
}
// W[K,O] (f32 row-major) -> Wt[O,K] (f16)
__global__ void cvt_transpose_f16(const float* __restrict__ W, _Float16* __restrict__ Wt,
                                  int K, int O) {
  int i = blockIdx.x * blockDim.x + threadIdx.x;
  if (i < K * O) {
    int k = i / O, o = i - k * O;
    Wt[(size_t)o * K + k] = (_Float16)W[i];
  }
}

// ---- self-loop attr (fill_value='mean') ----
__global__ void deg_kernel(const int* __restrict__ ei, float* __restrict__ deg) {
  int e = blockIdx.x * blockDim.x + threadIdx.x;
  if (e < N_EDGES) atomicAdd(&deg[ei[N_EDGES + e]], 1.0f);
}
__global__ void loopattr_sum(const int* __restrict__ ei, const float* __restrict__ ea,
                             float* __restrict__ la) {
  int i = blockIdx.x * blockDim.x + threadIdx.x;
  if (i < N_EDGES * EDIM) {
    int e = i >> 6, c = i & 63;
    atomicAdd(&la[(size_t)ei[N_EDGES + e] * EDIM + c], ea[i]);
  }
}
__global__ void loopattr_norm(float* __restrict__ la, const float* __restrict__ deg) {
  int i = blockIdx.x * blockDim.x + threadIdx.x;
  if (i < N_NODES * EDIM) la[i] /= fmaxf(deg[i >> 6], 1.0f);
}

// Ve[c,h] = sum_k We[c, h*HIDC+k] * a_edge[h,k]   (64x4, one block of 256)
__global__ void fold_aedge(const float* __restrict__ We, const float* __restrict__ ae,
                           float* __restrict__ Ve) {
  int i = threadIdx.x;           // 0..255
  int c = i >> 2, h = i & 3;
  float s = 0.f;
  for (int k = 0; k < HIDC; ++k) s += We[(size_t)c * HOUT + h * HIDC + k] * ae[h * HIDC + k];
  Ve[i] = s;
}

// asrc/adst[n,h] from xs[n,h,c]
__global__ void node_attn(const float* __restrict__ xs, const float* __restrict__ a_src,
                          const float* __restrict__ a_dst, float* __restrict__ asrc,
                          float* __restrict__ adst) {
  int i = blockIdx.x * blockDim.x + threadIdx.x;
  if (i < N_NODES * NHEAD) {
    int n = i >> 2, h = i & 3;
    const float* p = xs + (size_t)n * HOUT + h * HIDC;
    float s1 = 0.f, s2 = 0.f;
    for (int c = 0; c < HIDC; ++c) {
      float v = p[c];
      s1 += v * a_src[h * HIDC + c];
      s2 += v * a_dst[h * HIDC + c];
    }
    asrc[i] = s1; adst[i] = s2;
  }
}

// aedge[e,h] = ea_row(e) . Ve[:,h]
__global__ void edge_attn(const int* __restrict__ ei, const float* __restrict__ ea,
                          const float* __restrict__ la, const float* __restrict__ Ve,
                          float* __restrict__ buf) {
  int i = blockIdx.x * blockDim.x + threadIdx.x;
  if (i < ET * NHEAD) {
    int e = i >> 2, h = i & 3;
    const float* row = (e < N_EDGES) ? ea + (size_t)e * EDIM
                                     : la + (size_t)(e - N_EDGES) * EDIM;
    float s = 0.f;
    for (int c = 0; c < EDIM; ++c) s += row[c] * Ve[c * 4 + h];
    buf[i] = s;
  }
}

// logit = leaky(asrc[src]+adst[dst]+aedge); segment max into m
__global__ void logit_max(const int* __restrict__ ei, const float* __restrict__ asrc,
                          const float* __restrict__ adst, float* __restrict__ buf,
                          float* __restrict__ m) {
  int i = blockIdx.x * blockDim.x + threadIdx.x;
  if (i < ET * NHEAD) {
    int e = i >> 2, h = i & 3, s, d;
    if (e < N_EDGES) { s = ei[e]; d = ei[N_EDGES + e]; } else { s = d = e - N_EDGES; }
    float l = asrc[s * 4 + h] + adst[d * 4 + h] + buf[i];
    l = (l >= 0.f) ? l : 0.2f * l;
    buf[i] = l;
    atomicMaxFloat(&m[d * 4 + h], l);
  }
}
__global__ void exp_sum(const int* __restrict__ ei, const float* __restrict__ m,
                        float* __restrict__ buf, float* __restrict__ denom) {
  int i = blockIdx.x * blockDim.x + threadIdx.x;
  if (i < ET * NHEAD) {
    int e = i >> 2, h = i & 3;
    int d = (e < N_EDGES) ? ei[N_EDGES + e] : (e - N_EDGES);
    float p = __expf(buf[i] - m[d * 4 + h]);
    buf[i] = p;
    atomicAdd(&denom[d * 4 + h], p);
  }
}
__global__ void norm_alpha(const int* __restrict__ ei, const float* __restrict__ denom,
                           float* __restrict__ buf) {
  int i = blockIdx.x * blockDim.x + threadIdx.x;
  if (i < ET * NHEAD) {
    int e = i >> 2, h = i & 3;
    int d = (e < N_EDGES) ? ei[N_EDGES + e] : (e - N_EDGES);
    buf[i] = buf[i] / (denom[d * 4 + h] + 1e-16f) * 0.25f;  // fold head-mean 1/4
  }
}

// hb[dst,c] += alpha[e,h] * xs[src, h*128+c]   (grid-stride, Et*512 items)
__global__ void aggregate(const int* __restrict__ ei, const float* __restrict__ alpha,
                          const float* __restrict__ xs, float* __restrict__ hb) {
  long long total = (long long)ET * HOUT;
  for (long long i = (long long)blockIdx.x * blockDim.x + threadIdx.x; i < total;
       i += (long long)gridDim.x * blockDim.x) {
    int e = (int)(i >> 9);
    int c = (int)(i & 511);
    int h = c >> 7, cc = c & 127;
    int s, d;
    if (e < N_EDGES) { s = ei[e]; d = ei[N_EDGES + e]; } else { s = d = e - N_EDGES; }
    atomicAdd(&hb[(size_t)d * HIDC + cc], alpha[(size_t)e * 4 + h] * xs[(size_t)s * HOUT + c]);
  }
}
__global__ void bias_relu(float* __restrict__ h, const float* __restrict__ b) {
  int i = blockIdx.x * blockDim.x + threadIdx.x;
  if (i < N_NODES * HIDC) h[i] = fmaxf(h[i] + b[i & 127], 0.0f);
}

// ---- pooling + cat ----
__global__ void pool_cnt(const int* __restrict__ batch, float* __restrict__ cnt) {
  int n = blockIdx.x * blockDim.x + threadIdx.x;
  if (n < N_NODES) atomicAdd(&cnt[batch[n]], 1.0f);
}
__global__ void pool_sum(const int* __restrict__ batch, const float* __restrict__ h,
                         float* __restrict__ gs) {
  int i = blockIdx.x * blockDim.x + threadIdx.x;
  if (i < N_NODES * HIDC) atomicAdd(&gs[(size_t)batch[i >> 7] * HIDC + (i & 127)], h[i]);
}
__global__ void pool_div(float* __restrict__ g, const float* __restrict__ cnt) {
  int i = blockIdx.x * blockDim.x + threadIdx.x;
  if (i < N_BATCH * HIDC) g[i] /= fmaxf(cnt[i >> 7], 1.0f);
}
__global__ void make_cat(const float* __restrict__ h, const float* __restrict__ g,
                         const int* __restrict__ batch, _Float16* __restrict__ cat) {
  int i = blockIdx.x * blockDim.x + threadIdx.x;
  if (i < N_NODES * 2 * HIDC) {
    int n = i >> 8, c = i & 255;
    float v = (c < HIDC) ? h[(size_t)n * HIDC + c]
                         : g[(size_t)batch[n] * HIDC + (c - HIDC)];
    cat[i] = (_Float16)v;
  }
}
// values[n] = chid[n,:] . W2 + b2
__global__ void critic_out(const float* __restrict__ hid, const float* __restrict__ W2,
                           const float* __restrict__ b2, float* __restrict__ out) {
  int n = blockIdx.x * blockDim.x + threadIdx.x;
  if (n < N_NODES) {
    float s = b2[0];
    for (int c = 0; c < HIDC; ++c) s += hid[(size_t)n * HIDC + c] * W2[c];
    out[n] = s;
  }
}

// ---------------------------------------------------------------------------
static inline char* bump(char*& p, size_t bytes) {
  char* r = p;
  p += (bytes + 255) & ~(size_t)255;
  return r;
}
static inline int ceil_div(long long a, int b) { return (int)((a + b - 1) / b); }

extern "C" void kernel_launch(void* const* d_in, const int* in_sizes, int n_in,
                              void* d_out, int out_size, void* d_ws, size_t ws_size,
                              hipStream_t stream) {
  (void)in_sizes; (void)n_in; (void)out_size; (void)ws_size;
  const float* x      = (const float*)d_in[0];
  const int*   ei     = (const int*)d_in[1];
  const float* ea     = (const float*)d_in[2];
  const int*   batch  = (const int*)d_in[3];
  const float* g1W    = (const float*)d_in[4];
  const float* g1as   = (const float*)d_in[5];
  const float* g1ad   = (const float*)d_in[6];
  const float* g1ae   = (const float*)d_in[7];
  const float* g1We   = (const float*)d_in[8];
  const float* g1b    = (const float*)d_in[9];
  const float* g2W    = (const float*)d_in[10];
  const float* g2as   = (const float*)d_in[11];
  const float* g2ad   = (const float*)d_in[12];
  const float* g2ae   = (const float*)d_in[13];
  const float* g2We   = (const float*)d_in[14];
  const float* g2b    = (const float*)d_in[15];
  const float* aW1    = (const float*)d_in[16];
  const float* ab1    = (const float*)d_in[17];
  const float* aW2    = (const float*)d_in[18];
  const float* ab2    = (const float*)d_in[19];
  const float* cW1    = (const float*)d_in[20];
  const float* cb1    = (const float*)d_in[21];
  const float* cW2    = (const float*)d_in[22];
  const float* cb2    = (const float*)d_in[23];
  float* out_logits = (float*)d_out;                  // [N, NACT]
  float* out_values = (float*)d_out + (size_t)N_NODES * NACT;

  // ---- workspace layout ----
  char* wp = (char*)d_ws;
  float*    xs    = (float*)bump(wp, sizeof(float) * (size_t)N_NODES * HOUT);
  float*    hbuf  = (float*)bump(wp, sizeof(float) * (size_t)N_NODES * HIDC);
  float*    la    = (float*)bump(wp, sizeof(float) * (size_t)N_NODES * EDIM);
  float*    deg   = (float*)bump(wp, sizeof(float) * N_NODES);
  float*    asrc  = (float*)bump(wp, sizeof(float) * (size_t)N_NODES * NHEAD);
  float*    adst  = (float*)bump(wp, sizeof(float) * (size_t)N_NODES * NHEAD);
  float*    segm  = (float*)bump(wp, sizeof(float) * (size_t)N_NODES * NHEAD);
  float*    segd  = (float*)bump(wp, sizeof(float) * (size_t)N_NODES * NHEAD);
  float*    ebuf  = (float*)bump(wp, sizeof(float) * (size_t)ET * NHEAD);
  float*    ahid  = (float*)bump(wp, sizeof(float) * (size_t)N_NODES * HIDC);
  float*    chid  = (float*)bump(wp, sizeof(float) * (size_t)N_NODES * HIDC);
  float*    gpool = (float*)bump(wp, sizeof(float) * (size_t)N_BATCH * HIDC);
  float*    cnt   = (float*)bump(wp, sizeof(float) * N_BATCH);
  float*    Ve    = (float*)bump(wp, sizeof(float) * EDIM * NHEAD);
  _Float16* a16   = (_Float16*)bump(wp, sizeof(_Float16) * (size_t)N_NODES * HIDC);
  _Float16* cat16 = (_Float16*)bump(wp, sizeof(_Float16) * (size_t)N_NODES * 2 * HIDC);
  _Float16* gWt   = (_Float16*)bump(wp, sizeof(_Float16) * (size_t)HOUT * FIN);
  _Float16* aW1t  = (_Float16*)bump(wp, sizeof(_Float16) * (size_t)HIDC * 2 * HIDC);
  _Float16* aW2t  = (_Float16*)bump(wp, sizeof(_Float16) * (size_t)NACT * HIDC);
  _Float16* cW1t  = (_Float16*)bump(wp, sizeof(_Float16) * (size_t)HIDC * 2 * HIDC);

  const int T = 256;
  // ---- self-loop attrs (fill_value='mean') ----
  hipMemsetAsync(deg, 0, sizeof(float) * N_NODES, stream);
  hipMemsetAsync(la, 0, sizeof(float) * (size_t)N_NODES * EDIM, stream);
  deg_kernel<<<ceil_div(N_EDGES, T), T, 0, stream>>>(ei, deg);
  loopattr_sum<<<ceil_div((long long)N_EDGES * EDIM, T), T, 0, stream>>>(ei, ea, la);
  loopattr_norm<<<ceil_div((long long)N_NODES * EDIM, T), T, 0, stream>>>(la, deg);

  // x -> f16 A-matrix for layer 1
  cvt_f32_f16<<<ceil_div((long long)N_NODES * FIN, T), T, 0, stream>>>(x, a16, N_NODES * FIN);

  for (int layer = 0; layer < 2; ++layer) {
    const float* W   = layer ? g2W  : g1W;
    const float* as_ = layer ? g2as : g1as;
    const float* ad_ = layer ? g2ad : g1ad;
    const float* ae_ = layer ? g2ae : g1ae;
    const float* We  = layer ? g2We : g1We;
    const float* bb  = layer ? g2b  : g1b;

    // xs = A(f16) @ W  -> [N, 512] f32  (WMMA + LDS-staged weights)
    cvt_transpose_f16<<<ceil_div(FIN * HOUT, T), T, 0, stream>>>(W, gWt, FIN, HOUT);
    {
      dim3 grid(ceil_div(N_NODES, 128), HOUT / 64);
      wmma_gemm_lds<4, FIN><<<grid, 256, 0, stream>>>(a16, gWt, xs, nullptr,
                                                      N_NODES, HOUT, 0);
    }
    node_attn<<<ceil_div(N_NODES * NHEAD, T), T, 0, stream>>>(xs, as_, ad_, asrc, adst);
    fold_aedge<<<1, 256, 0, stream>>>(We, ae_, Ve);
    edge_attn<<<ceil_div((long long)ET * NHEAD, T), T, 0, stream>>>(ei, ea, la, Ve, ebuf);

    fill_f32<<<ceil_div(N_NODES * NHEAD, T), T, 0, stream>>>(segm, -3.0e38f, N_NODES * NHEAD);
    logit_max<<<ceil_div((long long)ET * NHEAD, T), T, 0, stream>>>(ei, asrc, adst, ebuf, segm);
    hipMemsetAsync(segd, 0, sizeof(float) * (size_t)N_NODES * NHEAD, stream);
    exp_sum<<<ceil_div((long long)ET * NHEAD, T), T, 0, stream>>>(ei, segm, ebuf, segd);
    norm_alpha<<<ceil_div((long long)ET * NHEAD, T), T, 0, stream>>>(ei, segd, ebuf);

    hipMemsetAsync(hbuf, 0, sizeof(float) * (size_t)N_NODES * HIDC, stream);
    aggregate<<<16384, 256, 0, stream>>>(ei, ebuf, xs, hbuf);
    bias_relu<<<ceil_div(N_NODES * HIDC, T), T, 0, stream>>>(hbuf, bb);

    // prepare next layer's A matrix
    cvt_f32_f16<<<ceil_div(N_NODES * HIDC, T), T, 0, stream>>>(hbuf, a16, N_NODES * HIDC);
  }

  // ---- global mean pool + broadcast cat ----
  hipMemsetAsync(gpool, 0, sizeof(float) * (size_t)N_BATCH * HIDC, stream);
  hipMemsetAsync(cnt, 0, sizeof(float) * N_BATCH, stream);
  pool_cnt<<<ceil_div(N_NODES, T), T, 0, stream>>>(batch, cnt);
  pool_sum<<<ceil_div(N_NODES * HIDC, T), T, 0, stream>>>(batch, hbuf, gpool);
  pool_div<<<ceil_div(N_BATCH * HIDC, T), T, 0, stream>>>(gpool, cnt);
  make_cat<<<ceil_div(N_NODES * 2 * HIDC, T), T, 0, stream>>>(hbuf, gpool, batch, cat16);

  // ---- actor / critic heads (WMMA GEMMs) ----
  cvt_transpose_f16<<<ceil_div(2 * HIDC * HIDC, T), T, 0, stream>>>(aW1, aW1t, 2 * HIDC, HIDC);
  cvt_transpose_f16<<<ceil_div(2 * HIDC * HIDC, T), T, 0, stream>>>(cW1, cW1t, 2 * HIDC, HIDC);
  {
    dim3 grid(ceil_div(N_NODES, 128), HIDC / 64);
    wmma_gemm_lds<4, 2 * HIDC><<<grid, 256, 0, stream>>>(cat16, aW1t, ahid, ab1,
                                                         N_NODES, HIDC, 1);
    wmma_gemm_lds<4, 2 * HIDC><<<grid, 256, 0, stream>>>(cat16, cW1t, chid, cb1,
                                                         N_NODES, HIDC, 1);
  }
  // logits = relu_hidden(f16) @ actor_W2 + b2  -> d_out   (2 N-tiles = 32 cols)
  cvt_f32_f16<<<ceil_div(N_NODES * HIDC, T), T, 0, stream>>>(ahid, a16, N_NODES * HIDC);
  cvt_transpose_f16<<<ceil_div(HIDC * NACT, T), T, 0, stream>>>(aW2, aW2t, HIDC, NACT);
  {
    dim3 grid(ceil_div(N_NODES, 128), NACT / 32);
    wmma_gemm_lds<2, HIDC><<<grid, 256, 0, stream>>>(a16, aW2t, out_logits, ab2,
                                                     N_NODES, NACT, 0);
  }
  // values: [N,128] . [128] + b
  critic_out<<<ceil_div(N_NODES, T), T, 0, stream>>>(chid, cW2, cb2, out_values);
}
